// NTA_Class_64209761075559
// MI455X (gfx1250) — matrix operand order repaired and checked by reading, not compiled, not run on hardware
//
#include <hip/hip_runtime.h>
#include <math.h>

#define CLS   2048
#define NN    4096   // 2*CLS rows of Pn / sim
#define BATCH 4096   // GEMM K dimension
#define KNOWN 1024
#define LDS_K 40     // padded bf16 row stride in LDS (80B -> 20-bank stride, conflict-free)

typedef __attribute__((ext_vector_type(16))) __bf16 v16bf;
typedef __attribute__((ext_vector_type(8)))  float  v8f;

union FragU { v16bf v; uint4 q[2]; };

static __device__ __forceinline__ unsigned short f2bf(float f) {
    unsigned u = __float_as_uint(f);
    u += 0x7FFFu + ((u >> 16) & 1u);   // round-to-nearest-even
    return (unsigned short)(u >> 16);
}
static __device__ __forceinline__ float bf2f(unsigned short h) {
    return __uint_as_float(((unsigned)h) << 16);
}
static __device__ __forceinline__ bool is_known(int i) {
    return (i < KNOWN) || (i >= CLS && i < CLS + KNOWN);
}

// ---------------- kernel 1: zero accumulators ----------------
__global__ void k_init(float* __restrict__ rowdenom, float* __restrict__ pos,
                       float* __restrict__ colsum, float* __restrict__ sumsq) {
    int i = blockIdx.x * blockDim.x + threadIdx.x;
    if (i < NN) { rowdenom[i] = 0.f; pos[i] = 0.f; colsum[i] = 0.f; sumsq[i] = 0.f; }
}

// ---------------- kernel 2: per-class sumsq + column sums (256 blocks, atomic combine) ----
__global__ void k_colstats(const float* __restrict__ prob, const float* __restrict__ prob_s,
                           float* __restrict__ sumsq, float* __restrict__ colsum) {
    int mat = blockIdx.z;                            // 0: prob, 1: prob_s
    int c   = blockIdx.x * 256 + threadIdx.x;        // class column 0..2047
    int r0  = blockIdx.y * (BATCH / 16);             // 16 row chunks of 256 rows
    const float* src = mat ? prob_s : prob;
    float s = 0.f, ss = 0.f;
    for (int r = r0; r < r0 + BATCH / 16; ++r) {
        float v = src[(size_t)r * CLS + c];
        s += v; ss += v * v;
    }
    int o = mat * CLS + c;
    atomicAdd(&colsum[o], s);
    atomicAdd(&sumsq[o], ss);
}

// ---------------- kernel 3: transpose + normalize + bf16 hi/lo split ----------------
__global__ void k_pack(const float* __restrict__ prob, const float* __restrict__ prob_s,
                       const float* __restrict__ sumsq,
                       unsigned short* __restrict__ hi, unsigned short* __restrict__ lo) {
    __shared__ float tile[32][33];
    int mat = blockIdx.z;
    const float* src = mat ? prob_s : prob;
    int c0 = blockIdx.x * 32, b0 = blockIdx.y * 32;
    #pragma unroll
    for (int yy = 0; yy < 4; ++yy) {
        int y = threadIdx.y + yy * 8;       // batch offset within tile
        tile[y][threadIdx.x] = src[(size_t)(b0 + y) * CLS + c0 + threadIdx.x];
    }
    __syncthreads();
    #pragma unroll
    for (int yy = 0; yy < 4; ++yy) {
        int y   = threadIdx.y + yy * 8;     // class offset within tile
        int cls = mat * CLS + c0 + y;       // global Pn row
        float inv = 1.f / fmaxf(sqrtf(sumsq[cls]), 1e-8f);
        float v = tile[threadIdx.x][y] * inv;
        unsigned short h = f2bf(v);
        float r = v - bf2f(h);
        size_t o = (size_t)cls * BATCH + b0 + threadIdx.x;
        hi[o] = h;
        lo[o] = f2bf(r);
    }
}

// ---------------- kernel 4: fused split-bf16 WMMA GEMM, double-buffered LDS ----------------
__global__ void __launch_bounds__(256) k_gemm(
        const unsigned short* __restrict__ Phi, const unsigned short* __restrict__ Plo,
        float* __restrict__ rowdenom, float* __restrict__ pos) {
    __shared__ __align__(16) unsigned short sAhi[2][128 * LDS_K];
    __shared__ __align__(16) unsigned short sAlo[2][128 * LDS_K];
    __shared__ __align__(16) unsigned short sBhi[2][128 * LDS_K];
    __shared__ __align__(16) unsigned short sBlo[2][128 * LDS_K];

    const int tid   = threadIdx.x;
    const int lane  = tid & 31;
    const int wid   = tid >> 5;          // 8 waves
    const int waveM = wid >> 1;          // 0..3  -> 32-row slab
    const int waveN = wid & 1;           // 0..1  -> 64-col slab
    const int m0 = blockIdx.y * 128, n0 = blockIdx.x * 128;

    // staging map: each thread owns 32B (two uint4) of one row of each 128x32 tile
    const int srow  = tid >> 1;          // 0..127
    const int skoff = (tid & 1) * 16;    // 0 or 16 (bf16 elems)
    const size_t gOffA = (size_t)(m0 + srow) * BATCH + skoff;
    const size_t gOffB = (size_t)(n0 + srow) * BATCH + skoff;
    const int lOff = srow * LDS_K + skoff;

    v8f acc[2][4];
    #pragma unroll
    for (int mt = 0; mt < 2; ++mt)
        #pragma unroll
        for (int nt = 0; nt < 4; ++nt)
            acc[mt][nt] = (v8f)(0.f);

    // prologue: stage k0 = 0 into buffer 0
    uint4 ah0 = *(const uint4*)&Phi[gOffA];
    uint4 ah1 = *(const uint4*)&Phi[gOffA + 8];
    uint4 al0 = *(const uint4*)&Plo[gOffA];
    uint4 al1 = *(const uint4*)&Plo[gOffA + 8];
    uint4 bh0 = *(const uint4*)&Phi[gOffB];
    uint4 bh1 = *(const uint4*)&Phi[gOffB + 8];
    uint4 bl0 = *(const uint4*)&Plo[gOffB];
    uint4 bl1 = *(const uint4*)&Plo[gOffB + 8];
    *(uint4*)&sAhi[0][lOff] = ah0;  *(uint4*)&sAhi[0][lOff + 8] = ah1;
    *(uint4*)&sAlo[0][lOff] = al0;  *(uint4*)&sAlo[0][lOff + 8] = al1;
    *(uint4*)&sBhi[0][lOff] = bh0;  *(uint4*)&sBhi[0][lOff + 8] = bh1;
    *(uint4*)&sBlo[0][lOff] = bl0;  *(uint4*)&sBlo[0][lOff + 8] = bl1;

    const int fr = lane & 15, kh = lane >> 4;

    #pragma unroll 2
    for (int it = 0; it < BATCH / 32; ++it) {
        const int cur = it & 1;
        const bool more = (it + 1) < (BATCH / 32);
        __syncthreads();

        // issue next tile's global loads early so they overlap the WMMA chain
        if (more) {
            size_t kadv = (size_t)(it + 1) * 32;
            ah0 = *(const uint4*)&Phi[gOffA + kadv];
            ah1 = *(const uint4*)&Phi[gOffA + kadv + 8];
            al0 = *(const uint4*)&Plo[gOffA + kadv];
            al1 = *(const uint4*)&Plo[gOffA + kadv + 8];
            bh0 = *(const uint4*)&Phi[gOffB + kadv];
            bh1 = *(const uint4*)&Phi[gOffB + kadv + 8];
            bl0 = *(const uint4*)&Plo[gOffB + kadv];
            bl1 = *(const uint4*)&Plo[gOffB + kadv + 8];
        }

        // fragment loads from current buffer
        v16bf fah[2], fal[2], fbh[4], fbl[4];
        #pragma unroll
        for (int mt = 0; mt < 2; ++mt) {
            // A frag, this code's k order: lane<16 holds K 0-15; lane>=16 holds K 16-31
            const unsigned short* p = &sAhi[cur][(waveM * 32 + mt * 16 + fr) * LDS_K];
            const unsigned short* q = &sAlo[cur][(waveM * 32 + mt * 16 + fr) * LDS_K];
            FragU f, g;
            f.q[0] = *(const uint4*)(p + (kh ? 16 : 0));
            f.q[1] = *(const uint4*)(p + (kh ? 24 : 8));
            g.q[0] = *(const uint4*)(q + (kh ? 16 : 0));
            g.q[1] = *(const uint4*)(q + (kh ? 24 : 8));
            fah[mt] = f.v; fal[mt] = g.v;
        }
        #pragma unroll
        for (int nt = 0; nt < 4; ++nt) {
            // B frag, this code's k order: lanes 0-15 hold K 0-15 of col n=lane; lanes 16-31 hold K 16-31
            const unsigned short* p = &sBhi[cur][(waveN * 64 + nt * 16 + fr) * LDS_K + (kh ? 16 : 0)];
            const unsigned short* q = &sBlo[cur][(waveN * 64 + nt * 16 + fr) * LDS_K + (kh ? 16 : 0)];
            FragU f, g;
            f.q[0] = *(const uint4*)(p);  f.q[1] = *(const uint4*)(p + 8);
            g.q[0] = *(const uint4*)(q);  g.q[1] = *(const uint4*)(q + 8);
            fbh[nt] = f.v; fbl[nt] = g.v;
        }

        // split-bf16: hi*hi + hi*lo + lo*hi (~fp32 accuracy), 24 WMMAs per K-step
        #pragma unroll
        for (int mt = 0; mt < 2; ++mt)
            #pragma unroll
            for (int nt = 0; nt < 4; ++nt) {
                acc[mt][nt] = __builtin_amdgcn_wmma_f32_16x16x32_bf16(
                    false, fah[mt], false, fbh[nt], (short)0, acc[mt][nt], false, false);
                acc[mt][nt] = __builtin_amdgcn_wmma_f32_16x16x32_bf16(
                    false, fah[mt], false, fbl[nt], (short)0, acc[mt][nt], false, false);
                acc[mt][nt] = __builtin_amdgcn_wmma_f32_16x16x32_bf16(
                    false, fal[mt], false, fbh[nt], (short)0, acc[mt][nt], false, false);
            }

        // write next tile into the other buffer (its readers finished before this
        // iteration's barrier, so no extra barrier is needed here)
        if (more) {
            const int nxt = 1 - cur;
            *(uint4*)&sAhi[nxt][lOff] = ah0;  *(uint4*)&sAhi[nxt][lOff + 8] = ah1;
            *(uint4*)&sAlo[nxt][lOff] = al0;  *(uint4*)&sAlo[nxt][lOff + 8] = al1;
            *(uint4*)&sBhi[nxt][lOff] = bh0;  *(uint4*)&sBhi[nxt][lOff + 8] = bh1;
            *(uint4*)&sBlo[nxt][lOff] = bl0;  *(uint4*)&sBlo[nxt][lOff + 8] = bl1;
        }
    }

    // epilogue: apply contrastive masking rules, accumulate exp-sums per row.
    // C/D layout: vgpr v, lane l -> row M = v + 8*(l>>4), col N = l&15
    const int cr = lane & 15, ch = lane >> 4;
    #pragma unroll
    for (int mt = 0; mt < 2; ++mt) {
        #pragma unroll
        for (int v = 0; v < 8; ++v) {
            int i = m0 + waveM * 32 + mt * 16 + v + 8 * ch;
            int partner = (i < CLS) ? (i + CLS) : (i - CLS);
            bool known = is_known(i);
            float local = 0.f;
            #pragma unroll
            for (int nt = 0; nt < 4; ++nt) {
                int j = n0 + waveN * 64 + nt * 16 + cr;
                float s = acc[mt][nt][v] * 2.0f;          // / TEMP (0.5)
                if (j == partner) {
                    pos[i] = s;                           // unique writer
                } else if (j != i && !(known && j < KNOWN)) {
                    local += expf(s);                     // zeroed negs counted analytically later
                }
            }
            // lanes 0-15 (and 16-31) share the same row i -> reduce within 16-lane group
            #pragma unroll
            for (int m = 8; m >= 1; m >>= 1)
                local += __shfl_xor(local, m, 32);
            if (cr == 0) atomicAdd(&rowdenom[i], local);
        }
    }
}

// ---------------- kernel 5: final scalar ----------------
__global__ void k_finalize(const float* __restrict__ rowdenom, const float* __restrict__ pos,
                           const float* __restrict__ colsum, float* __restrict__ out) {
    __shared__ float red[256];
    int tid = threadIdx.x;
    float ce = 0.f;
    for (int i = tid; i < NN; i += 256) {
        float extra = is_known(i) ? (float)(KNOWN - 1) : 0.f;  // zeroed negatives: exp(0)=1 each
        float p = pos[i];
        ce += logf(rowdenom[i] + expf(p) + extra) - p;
    }
    float s1 = 0.f, sl1 = 0.f, s2 = 0.f, sl2 = 0.f;
    for (int c = tid; c < CLS; c += 256) {
        float a = colsum[c];        s1 += a; sl1 += a * logf(a);
        float b = colsum[CLS + c];  s2 += b; sl2 += b * logf(b);
    }
    float vals[5] = {ce, s1, sl1, s2, sl2};
    float tot[5];
    #pragma unroll
    for (int t = 0; t < 5; ++t) {
        __syncthreads();
        red[tid] = vals[t];
        __syncthreads();
        for (int step = 128; step >= 1; step >>= 1) {
            if (tid < step) red[tid] += red[tid + step];
            __syncthreads();
        }
        tot[t] = red[0];
    }
    if (tid == 0) {
        float logC = logf((float)CLS);
        // sum m*log m = (sum c*log c)/T - log T   with m = c/T
        float reg = 2.f * logC + (tot[2] / tot[1] - logf(tot[1]))
                               + (tot[4] / tot[3] - logf(tot[3]));
        out[0] = tot[0] / (float)NN + reg;
    }
}

extern "C" void kernel_launch(void* const* d_in, const int* in_sizes, int n_in,
                              void* d_out, int out_size, void* d_ws, size_t ws_size,
                              hipStream_t stream) {
    (void)in_sizes; (void)n_in; (void)out_size; (void)ws_size;
    const float* prob   = (const float*)d_in[0];
    const float* prob_s = (const float*)d_in[1];

    char* ws = (char*)d_ws;
    float* sumsq    = (float*)(ws);
    float* colsum   = (float*)(ws + 16 * 1024);
    float* rowdenom = (float*)(ws + 32 * 1024);
    float* pos      = (float*)(ws + 48 * 1024);
    unsigned short* Phi = (unsigned short*)(ws + 64 * 1024);
    unsigned short* Plo = Phi + (size_t)NN * BATCH;     // 32MB each plane
    float* out = (float*)d_out;

    k_init<<<16, 256, 0, stream>>>(rowdenom, pos, colsum, sumsq);
    k_colstats<<<dim3(CLS / 256, 16, 2), 256, 0, stream>>>(prob, prob_s, sumsq, colsum);
    k_pack<<<dim3(CLS / 32, BATCH / 32, 2), dim3(32, 8), 0, stream>>>(prob, prob_s, sumsq, Phi, Plo);
    k_gemm<<<dim3(NN / 128, NN / 128), 256, 0, stream>>>(Phi, Plo, rowdenom, pos);
    k_finalize<<<1, 256, 0, stream>>>(rowdenom, pos, colsum, out);
}
